// FeedForwardNetwork_29927332118744
// MI455X (gfx1250) — matrix operand order repaired and checked by reading, not compiled, not run on hardware
//
#include <hip/hip_runtime.h>

// ---------------------------------------------------------------------------
// Problem constants (from the reference)
// ---------------------------------------------------------------------------
#define N_NODES   65536
#define X_COLS    480
#define M0_IN     128
#define M1_IN     64
#define M2_IN     32
#define M0_MID    384
#define M1_MID    192
#define M2_MID    96
#define H0_OUT    672   // M0_MID + M1_MID + M2_MID

// Weight workspace offsets (bf16 elements), fragment-swizzled order
#define OFF_W1_0  0
#define OFF_W1_1  86016   // 128*672
#define OFF_W1_2  98304   // + 64*192
#define OFF_W2_0  101376  // + 32*96
#define OFF_W2_1  150528  // + 384*128
#define OFF_W2_2  162816  // + 192*64
#define WQ_TOTAL  165888  // + 96*32

typedef __attribute__((ext_vector_type(16))) __bf16 v16bf;
typedef __attribute__((ext_vector_type(8)))  float  v8f;
static_assert(sizeof(v16bf) == 32, "v16bf must be 32B");

#define MT 2     // row-tiles (of 16 nodes) per wave: amortizes B-fragment loads

// ---------------------------------------------------------------------------
// WMMA wrapper: D = A(16x32 bf16) * B(32x16 bf16) + C(16x16 f32)
// ---------------------------------------------------------------------------
__device__ __forceinline__ v8f wmma_bf16(v16bf a, v16bf b, v8f c) {
  return __builtin_amdgcn_wmma_f32_16x16x32_bf16(
      /*neg_a=*/false, a, /*neg_b=*/false, b,
      /*c_mod=*/(short)0, c, /*reuse_a=*/false, /*reuse_b=*/false);
}

// Fast sigmoid: single v_exp_f32 + v_rcp_f32 (no IEEE div expansion)
__device__ __forceinline__ float sigf(float v) {
  return __builtin_amdgcn_rcpf(1.0f + __expf(-v));
}

// B fragment: pre-swizzled in ws; one 32B load per lane.
__device__ __forceinline__ v16bf load_b_frag(const __bf16* __restrict__ w,
                                             int frag, int lane) {
  return *(const v16bf*)(w + (size_t)frag * 512 + lane * 16);
}

// A fragment from a dense fp32 row (contiguous K), p points at (row, k0).
__device__ __forceinline__ v16bf a_frag_dense(const float* __restrict__ p) {
  float4 q0 = ((const float4*)p)[0];
  float4 q1 = ((const float4*)p)[1];
  float4 q2 = ((const float4*)(p + 16))[0];
  float4 q3 = ((const float4*)(p + 16))[1];
  v16bf a;
  a[0]=(__bf16)q0.x; a[1]=(__bf16)q0.y; a[2]=(__bf16)q0.z;  a[3]=(__bf16)q0.w;
  a[4]=(__bf16)q1.x; a[5]=(__bf16)q1.y; a[6]=(__bf16)q1.z;  a[7]=(__bf16)q1.w;
  a[8]=(__bf16)q2.x; a[9]=(__bf16)q2.y; a[10]=(__bf16)q2.z; a[11]=(__bf16)q2.w;
  a[12]=(__bf16)q3.x;a[13]=(__bf16)q3.y;a[14]=(__bf16)q3.z; a[15]=(__bf16)q3.w;
  return a;
}

// A fragment from a channel-strided view: col = colbase + stride*K
__device__ __forceinline__ v16bf a_frag_ch(const float* __restrict__ xrow,
                                           int colbase, int stride, int k0) {
  v16bf a;
#pragma unroll
  for (int j = 0; j < 8; ++j) a[j]     = (__bf16)xrow[colbase + (k0 + j) * stride];
#pragma unroll
  for (int j = 0; j < 8; ++j) a[8 + j] = (__bf16)xrow[colbase + (k0 + 16 + j) * stride];
  return a;
}

// A fragment from an LDS bf16 row-major [16 x C] buffer (two b128 ds loads).
__device__ __forceinline__ v16bf a_frag_lds(const __bf16* buf, int row, int C, int k0) {
  union { uint4 q[2]; v16bf v; } u;
  u.q[0] = *(const uint4*)(buf + row * C + k0);
  u.q[1] = *(const uint4*)(buf + row * C + k0 + 16);
  return u.v;
}

// ---------------------------------------------------------------------------
// Prep: convert fp32 weight [K x Nout] row-major into bf16 WMMA-B fragments.
// frag(nt,kc) = block of 512 bf16: idx = lane*16+elem,
//   K = kc*32 + (elem<8 ? 8*(lane>>4)+elem : 16+8*(lane>>4)+elem-8), N = nt*16 + (lane&15)
// ---------------------------------------------------------------------------
__global__ void prep_weights(const float* __restrict__ src, __bf16* __restrict__ dst,
                             int K, int Nout) {
  int total = K * Nout;
  int nkc = K >> 5;
  for (int d = blockIdx.x * blockDim.x + threadIdx.x; d < total;
       d += gridDim.x * blockDim.x) {
    int blk = d >> 9, within = d & 511;
    int lane = within >> 4, elem = within & 15;
    int kc = blk % nkc, nt = blk / nkc;
    int k = kc * 32 + ((elem < 8) ? (8 * (lane >> 4) + elem) : (16 + 8 * (lane >> 4) + elem - 8));
    int n = nt * 16 + (lane & 15);
    dst[d] = (__bf16)src[k * Nout + n];
  }
}

// ---------------------------------------------------------------------------
// Main fused FFN kernel: one wave per block; the wave owns MT=2 row-tiles
// (32 nodes).  Each B fragment is loaded once and fed to MT independent
// WMMA accumulation chains.
// ---------------------------------------------------------------------------
__global__ __launch_bounds__(32)
void ffn_kernel(const float* __restrict__ x,
                const __bf16* __restrict__ wq,
                const float* __restrict__ b1_0,
                const float* __restrict__ b2_0,
                float* __restrict__ out) {
  __shared__ __align__(16) __bf16 lds_scal[MT][16 * M0_MID];
  __shared__ __align__(16) __bf16 lds_g1 [MT][16 * M1_MID];
  __shared__ __align__(16) __bf16 lds_g2 [MT][16 * M2_MID];
  __shared__ __align__(16) __bf16 lds_h  [MT][16 * M1_MID];

  const int lane  = threadIdx.x & 31;
  const int rowb  = blockIdx.x * (16 * MT);           // first node of this wave
  const int ncol  = lane & 15;                        // D-layout column
  const int rbase = (lane < 16) ? 0 : 8;              // D-layout row offset
  const int kb    = (lane < 16) ? 0 : 8;              // A-layout K offset

  const float* xA[MT];
#pragma unroll
  for (int t = 0; t < MT; ++t)
    xA[t] = x + (size_t)(rowb + t * 16 + ncol) * X_COLS;

  const __bf16* W1_0 = wq + OFF_W1_0;
  const __bf16* W1_1 = wq + OFF_W1_1;
  const __bf16* W1_2 = wq + OFF_W1_2;
  const __bf16* W2_0 = wq + OFF_W2_0;
  const __bf16* W2_1 = wq + OFF_W2_1;
  const __bf16* W2_2 = wq + OFF_W2_2;

  // ---- Stage A-frags of s = x[:, 0:128] (K=128 -> 4 chunks) per tile
  v16bf aS[MT][4];
#pragma unroll
  for (int t = 0; t < MT; ++t)
#pragma unroll
    for (int kc = 0; kc < 4; ++kc)
      aS[t][kc] = a_frag_dense(xA[t] + kc * 32 + kb);

  // ================= h0 = s @ w1_0 + b1_0 ; activations -> LDS ============
#pragma unroll 1
  for (int nt = 0; nt < H0_OUT / 16; ++nt) {   // 42 tiles
    v8f acc[MT] = {};
#pragma unroll
    for (int kc = 0; kc < 4; ++kc) {
      v16bf b = load_b_frag(W1_0, nt * 4 + kc, lane);
#pragma unroll
      for (int t = 0; t < MT; ++t) acc[t] = wmma_bf16(aS[t][kc], b, acc[t]);
    }
    float bias = b1_0[nt * 16 + ncol];
#pragma unroll
    for (int t = 0; t < MT; ++t) {
      if (nt < 24) {                   // scal = silu(h0)
#pragma unroll
        for (int r = 0; r < 8; ++r) {
          float v = acc[t][r] + bias;
          lds_scal[t][(rbase + r) * M0_MID + nt * 16 + ncol] = (__bf16)(v * sigf(v));
        }
      } else if (nt < 36) {            // g1 = sigmoid
#pragma unroll
        for (int r = 0; r < 8; ++r) {
          float v = acc[t][r] + bias;
          lds_g1[t][(rbase + r) * M1_MID + (nt - 24) * 16 + ncol] = (__bf16)sigf(v);
        }
      } else {                         // g2 = sigmoid
#pragma unroll
        for (int r = 0; r < 8; ++r) {
          float v = acc[t][r] + bias;
          lds_g2[t][(rbase + r) * M2_MID + (nt - 36) * 16 + ncol] = (__bf16)sigf(v);
        }
      }
    }
  }
  __syncthreads();

  // ================= o0 = scal @ w2_0 + b2_0 -> out[:, 0:128] =============
#pragma unroll 1
  for (int nt = 0; nt < M0_IN / 16; ++nt) {    // 8 tiles
    v8f acc[MT] = {};
#pragma unroll 1
    for (int kc = 0; kc < M0_MID / 32; ++kc) { // 12 chunks
      v16bf b = load_b_frag(W2_0, nt * 12 + kc, lane);
#pragma unroll
      for (int t = 0; t < MT; ++t) {
        v16bf a = a_frag_lds(lds_scal[t], ncol, M0_MID, kc * 32 + kb);
        acc[t] = wmma_bf16(a, b, acc[t]);
      }
    }
    float bias = b2_0[nt * 16 + ncol];
#pragma unroll
    for (int t = 0; t < MT; ++t)
#pragma unroll
      for (int r = 0; r < 8; ++r)
        out[(size_t)(rowb + t * 16 + rbase + r) * X_COLS + nt * 16 + ncol]
            = acc[t][r] + bias;
  }

  // ================= v1 channels: (g1 * (v1c@w1_1)) @ w2_1 ================
#pragma unroll 1
  for (int c = 0; c < 3; ++c) {
    v16bf aV[MT][2];
#pragma unroll
    for (int t = 0; t < MT; ++t)
#pragma unroll
      for (int kc = 0; kc < 2; ++kc)
        aV[t][kc] = a_frag_ch(xA[t], M0_IN + c, 3, kc * 32 + kb);
    // GEMM1 (64 -> 192), gate, stash bf16 in LDS
#pragma unroll 1
    for (int nt = 0; nt < M1_MID / 16; ++nt) { // 12 tiles
      v8f acc[MT] = {};
#pragma unroll
      for (int kc = 0; kc < 2; ++kc) {
        v16bf b = load_b_frag(W1_1, nt * 2 + kc, lane);
#pragma unroll
        for (int t = 0; t < MT; ++t) acc[t] = wmma_bf16(aV[t][kc], b, acc[t]);
      }
#pragma unroll
      for (int t = 0; t < MT; ++t)
#pragma unroll
        for (int r = 0; r < 8; ++r) {
          int lrow = rbase + r, col = nt * 16 + ncol;
          float g = (float)lds_g1[t][lrow * M1_MID + col];
          lds_h[t][lrow * M1_MID + col] = (__bf16)(acc[t][r] * g);
        }
    }
    __syncthreads();
    // GEMM2 (192 -> 64) -> out[:, 128 + 3*o + c]
#pragma unroll 1
    for (int nt = 0; nt < M1_IN / 16; ++nt) {  // 4 tiles
      v8f acc[MT] = {};
#pragma unroll 1
      for (int kc = 0; kc < M1_MID / 32; ++kc) { // 6 chunks
        v16bf b = load_b_frag(W2_1, nt * 6 + kc, lane);
#pragma unroll
        for (int t = 0; t < MT; ++t) {
          v16bf a = a_frag_lds(lds_h[t], ncol, M1_MID, kc * 32 + kb);
          acc[t] = wmma_bf16(a, b, acc[t]);
        }
      }
#pragma unroll
      for (int t = 0; t < MT; ++t)
#pragma unroll
        for (int r = 0; r < 8; ++r) {
          int o = nt * 16 + ncol;
          out[(size_t)(rowb + t * 16 + rbase + r) * X_COLS + M0_IN + 3 * o + c]
              = acc[t][r];
        }
    }
    __syncthreads();
  }

  // ================= v2 channels: (g2 * (v2c@w1_2)) @ w2_2 ================
#pragma unroll 1
  for (int c = 0; c < 5; ++c) {
    v16bf aV[MT];
#pragma unroll
    for (int t = 0; t < MT; ++t)
      aV[t] = a_frag_ch(xA[t], M0_IN + 3 * M1_IN + c, 5, kb);  // K=32, 1 chunk
    // GEMM1 (32 -> 96), gate, stash
#pragma unroll 1
    for (int nt = 0; nt < M2_MID / 16; ++nt) { // 6 tiles
      v16bf b = load_b_frag(W1_2, nt, lane);
      v8f acc[MT];
#pragma unroll
      for (int t = 0; t < MT; ++t) {
        v8f z = {};
        acc[t] = wmma_bf16(aV[t], b, z);
      }
#pragma unroll
      for (int t = 0; t < MT; ++t)
#pragma unroll
        for (int r = 0; r < 8; ++r) {
          int lrow = rbase + r, col = nt * 16 + ncol;
          float g = (float)lds_g2[t][lrow * M2_MID + col];
          lds_h[t][lrow * M2_MID + col] = (__bf16)(acc[t][r] * g);
        }
    }
    __syncthreads();
    // GEMM2 (96 -> 32) -> out[:, 320 + 5*o + c]
#pragma unroll 1
    for (int nt = 0; nt < M2_IN / 16; ++nt) {  // 2 tiles
      v8f acc[MT] = {};
#pragma unroll 1
      for (int kc = 0; kc < M2_MID / 32; ++kc) { // 3 chunks
        v16bf b = load_b_frag(W2_2, nt * 3 + kc, lane);
#pragma unroll
        for (int t = 0; t < MT; ++t) {
          v16bf a = a_frag_lds(lds_h[t], ncol, M2_MID, kc * 32 + kb);
          acc[t] = wmma_bf16(a, b, acc[t]);
        }
      }
#pragma unroll
      for (int t = 0; t < MT; ++t)
#pragma unroll
        for (int r = 0; r < 8; ++r) {
          int o = nt * 16 + ncol;
          out[(size_t)(rowb + t * 16 + rbase + r) * X_COLS + M0_IN + 3 * M1_IN + 5 * o + c]
              = acc[t][r];
        }
    }
    __syncthreads();
  }
}

// ---------------------------------------------------------------------------
// Launch
// ---------------------------------------------------------------------------
extern "C" void kernel_launch(void* const* d_in, const int* in_sizes, int n_in,
                              void* d_out, int out_size, void* d_ws, size_t ws_size,
                              hipStream_t stream) {
  const float* x    = (const float*)d_in[0];
  const float* w1_0 = (const float*)d_in[1];
  const float* b1_0 = (const float*)d_in[2];
  const float* w1_1 = (const float*)d_in[3];
  const float* w1_2 = (const float*)d_in[4];
  const float* w2_0 = (const float*)d_in[5];
  const float* b2_0 = (const float*)d_in[6];
  const float* w2_1 = (const float*)d_in[7];
  const float* w2_2 = (const float*)d_in[8];
  float* out  = (float*)d_out;
  __bf16* wq  = (__bf16*)d_ws;   // needs WQ_TOTAL*2 = ~332 KB of workspace

  auto prep = [&](const float* src, int off, int K, int Nout) {
    int total = K * Nout;
    int blocks = (total + 255) / 256;
    prep_weights<<<blocks, 256, 0, stream>>>(src, wq + off, K, Nout);
  };
  prep(w1_0, OFF_W1_0, M0_IN,  H0_OUT);
  prep(w1_1, OFF_W1_1, M1_IN,  M1_MID);
  prep(w1_2, OFF_W1_2, M2_IN,  M2_MID);
  prep(w2_0, OFF_W2_0, M0_MID, M0_IN);
  prep(w2_1, OFF_W2_1, M1_MID, M1_IN);
  prep(w2_2, OFF_W2_2, M2_MID, M2_IN);

  // 65536 nodes / (16 * MT per wave) = 2048 blocks of one wave
  ffn_kernel<<<N_NODES / (16 * MT), 32, 0, stream>>>(x, wq, b1_0, b2_0, out);
}
